// TemporalLSTM_2894807957763
// MI455X (gfx1250) — matrix operand order repaired and checked by reading, not compiled, not run on hardware
//
#include <hip/hip_runtime.h>

#define TSEQ 200
#define NIN  5
#define HID  32
#define NB   16384

typedef __attribute__((ext_vector_type(16))) _Float16 v16h;
typedef __attribute__((ext_vector_type(8)))  float    v8f;

union V16H {
  v16h v;
  unsigned int u[8];
  _Float16 h[16];
};

// pack two f32 -> packed f16 pair (v_cvt_pk_f16_f32)
static __device__ __forceinline__ unsigned int pk16(float a, float b) {
  union { _Float16 q[2]; unsigned int u; } c;
  c.q[0] = (_Float16)a;
  c.q[1] = (_Float16)b;
  return c.u;
}

// gfx1250 has V_TANH_F32 -- 1 transcendental op for tanh, and sigmoid via
// sig(x) = 0.5*tanh(0.5x)+0.5 (mul + tanh + fma). Fall back to exp2-based
// forms if this toolchain lacks the builtin.
#if __has_builtin(__builtin_amdgcn_tanhf)
static __device__ __forceinline__ float ftanh_(float x) {
  return __builtin_amdgcn_tanhf(x);
}
static __device__ __forceinline__ float fsig(float x) {
  return __builtin_fmaf(0.5f, __builtin_amdgcn_tanhf(0.5f * x), 0.5f);
}
#else
static __device__ __forceinline__ float fsig(float x) {
  return __builtin_amdgcn_rcpf(1.0f + __expf(-x));
}
static __device__ __forceinline__ float ftanh_(float x) {
  return 2.0f * __builtin_amdgcn_rcpf(1.0f + __expf(-2.0f * x)) - 1.0f;
}
#endif

__global__ __launch_bounds__(256, 1)
void lstm_fused(const float* __restrict__ x,
                const float* __restrict__ W_ih,
                const float* __restrict__ W_hh,
                const float* __restrict__ b_ih,
                const float* __restrict__ b_hh,
                const float* __restrict__ W_fc,
                const float* __restrict__ b_fc,
                float* __restrict__ out)
{
  // per-wave 16x32 f16 h-tile, row stride 40 halfs (80 B) -> conflict-free b128 reads
  __shared__ _Float16 hbuf[8][16 * 40];

  const int wave    = threadIdx.x >> 5;
  const int lane    = threadIdx.x & 31;
  const int Lm      = lane & 15;
  const int hiHalf  = lane >> 4;            // 0: lanes 0-15, 1: lanes 16-31
  const int rowBase = (blockIdx.x * 8 + wave) * 16;

  // ---- build loop-invariant B tiles in registers (f16) ----
  // gate-column permutation: tile t, column Lm -> gate G = 32*(t/2) + 2*Lm + (t&1)
  // so each lane's C fragment holds adjacent hidden units (2Lm, 2Lm+1).
  v16h Bhh[8], Bih[8];
#pragma unroll
  for (int t = 0; t < 8; ++t) {
    const int G  = 32 * (t >> 1) + 2 * Lm + (t & 1);
    const int k0 = hiHalf * 16;             // B k order used here (same as A): lanes<16 hold K 0..15, lanes>=16 hold K 16..31
    V16H bh;
    const float* wr = W_hh + G * HID + k0;
#pragma unroll
    for (int j = 0; j < 8; ++j) bh.u[j] = pk16(wr[2 * j], wr[2 * j + 1]);
    Bhh[t] = bh.v;

    V16H bi;
#pragma unroll
    for (int j = 0; j < 8; ++j) bi.u[j] = 0u;
    if (hiHalf == 0) {                      // x-projection only uses K=0..5
      const float* wi  = W_ih + G * NIN;
      const float bias = b_ih[G] + b_hh[G];
      bi.u[0] = pk16(wi[0], wi[1]);
      bi.u[1] = pk16(wi[2], wi[3]);
      bi.u[2] = pk16(wi[4], bias);          // K=5 row carries bias; A_x K=5 element is 1.0
    }
    Bih[t] = bi.v;
  }

  // ---- zero h tile (h0 = 0), c = 0 ----
  _Float16* hb = &hbuf[wave][0];
#pragma unroll
  for (int r = 0; r < 8; ++r)
    *(unsigned int*)(hb + (hiHalf * 8 + r) * 40 + 2 * Lm) = 0u;

  float ca[8], cb[8];
#pragma unroll
  for (int r = 0; r < 8; ++r) { ca[r] = 0.0f; cb[r] = 0.0f; }

  // A operand, this code's contiguous k order: lanes<16 hold row Lm, K 0..15; lanes>=16 K 16..31
  const _Float16* aload = hb + Lm * 40 + hiHalf * 16;
  const float*    xr    = x + (size_t)(rowBase + Lm) * (TSEQ * NIN);
  const float     m     = hiHalf ? 0.0f : 1.0f;   // zero the upper K-half of A_x without diverging

  // software-pipelined x fetch: xv holds x[t] while x[t+1] streams in
  float xv[NIN];
#pragma unroll
  for (int i = 0; i < NIN; ++i) xv[i] = xr[i];

  for (int t = 0; t < TSEQ; ++t) {
    if (t + 8 < TSEQ) __builtin_prefetch(xr + (t + 8) * NIN, 0, 0); // global_prefetch_b8

    // ---- A_x: x[t] in K=0..4, constant 1.0 in K=5 (bias row), zeros elsewhere ----
    V16H ax;
#pragma unroll
    for (int j = 0; j < 8; ++j) ax.u[j] = 0u;
    ax.u[0] = pk16(xv[0] * m, xv[1] * m);
    ax.u[1] = pk16(xv[2] * m, xv[3] * m);
    ax.u[2] = pk16(xv[4] * m, m);

    // issue next step's x loads now so their latency hides under WMMA+activations
    if (t + 1 < TSEQ) {
      const float* xn = xr + (t + 1) * NIN;
#pragma unroll
      for (int i = 0; i < NIN; ++i) xv[i] = xn[i];
    }

    // ---- A_h: previous h from LDS (two ds_load_b128) ----
    V16H ah;
    {
      const uint4 lo = *(const uint4*)(aload);
      const uint4 hi = *(const uint4*)(aload + 8);
      ah.u[0] = lo.x; ah.u[1] = lo.y; ah.u[2] = lo.z; ah.u[3] = lo.w;
      ah.u[4] = hi.x; ah.u[5] = hi.y; ah.u[6] = hi.z; ah.u[7] = hi.w;
    }

    // ---- gates = x@W_ih^T + bias + h@W_hh^T : two chained WMMAs per 16-gate tile ----
    v8f acc[8];
#pragma unroll
    for (int q = 0; q < 8; ++q) {
      v8f z = {0.f, 0.f, 0.f, 0.f, 0.f, 0.f, 0.f, 0.f};
      acc[q] = __builtin_amdgcn_wmma_f32_16x16x32_f16(false, ax.v, false, Bih[q],
                                                      (short)0, z, false, false);
      acc[q] = __builtin_amdgcn_wmma_f32_16x16x32_f16(false, ah.v, false, Bhh[q],
                                                      (short)0, acc[q], false, false);
    }

    // ---- elementwise cell update; lane owns rows hiHalf*8+r, units (2Lm, 2Lm+1) ----
#pragma unroll
    for (int r = 0; r < 8; ++r) {
      const float ia = acc[0][r], ib = acc[1][r];
      const float fa = acc[2][r], fb = acc[3][r];
      const float ga = acc[4][r], gb = acc[5][r];
      const float oa = acc[6][r], ob = acc[7][r];
      ca[r] = fsig(fa) * ca[r] + fsig(ia) * ftanh_(ga);
      cb[r] = fsig(fb) * cb[r] + fsig(ib) * ftanh_(gb);
      const float ha  = fsig(oa) * ftanh_(ca[r]);
      const float hbv = fsig(ob) * ftanh_(cb[r]);
      // packed write of adjacent units -> single ds_store_b32
      *(unsigned int*)(hb + (hiHalf * 8 + r) * 40 + 2 * Lm) = pk16(ha, hbv);
    }
  }

  // ---- final FC: out[row, j] = b_fc[j] + sum_n h[row,n] * W_fc[j,n]
  // lane handles (row = Lm, j = hiHalf): all 32 lanes busy, coalesced store.
  const _Float16* hr = hb + Lm * 40;
  float accf = b_fc[hiHalf];
  const float* wf = W_fc + hiHalf * HID;
#pragma unroll
  for (int n = 0; n < HID; ++n)
    accf += (float)hr[n] * wf[n];
  out[(size_t)(rowBase + Lm) * 2 + hiHalf] = accf;
}

extern "C" void kernel_launch(void* const* d_in, const int* in_sizes, int n_in,
                              void* d_out, int out_size, void* d_ws, size_t ws_size,
                              hipStream_t stream) {
  const float* x    = (const float*)d_in[0];
  const float* W_ih = (const float*)d_in[1];
  const float* W_hh = (const float*)d_in[2];
  const float* b_ih = (const float*)d_in[3];
  const float* b_hh = (const float*)d_in[4];
  const float* W_fc = (const float*)d_in[5];
  const float* b_fc = (const float*)d_in[6];
  float* out = (float*)d_out;

  // 16384 rows / (8 waves * 16 rows) = 128 blocks of 256 threads (8 wave32)
  lstm_fused<<<dim3(NB / 128), dim3(256), 0, stream>>>(
      x, W_ih, W_hh, b_ih, b_hh, W_fc, b_fc, out);
}
